// PositionLinearAttention_61186104099758
// MI455X (gfx1250) — compile-verified, matrix operand order checked by reading, not run on hardware
//
#include <hip/hip_runtime.h>
#include <hip/hip_bf16.h>
#include <math.h>

typedef __bf16 bf16;
typedef __attribute__((ext_vector_type(16))) __bf16 v16bf;
typedef __attribute__((ext_vector_type(8)))  float  v8f;
typedef __attribute__((ext_vector_type(4)))  int    v4i;

static constexpr int BSZ = 16;    // batch
static constexpr int CDIM = 2048; // channels
static constexpr int DDIM = 1024; // d_model == L
static constexpr int BDIM = 256;  // threads per block (8 waves)

// ---------------------------------------------------------------------------
// CDNA5 async global->LDS copy (ASYNCcnt-tracked, no VGPR data registers)
// builtin signature (from hipcc diagnostic): (v4i as1*, v4i as3*, Imm, Imm)
// ---------------------------------------------------------------------------
#if defined(__has_builtin)
#if __has_builtin(__builtin_amdgcn_global_load_async_to_lds_b128)
#define HAS_ASYNC_LDS 1
#endif
#endif
#ifndef HAS_ASYNC_LDS
#define HAS_ASYNC_LDS 0
#endif

#if HAS_ASYNC_LDS
typedef __attribute__((address_space(1))) v4i gv4i;
typedef __attribute__((address_space(3))) v4i lv4i;
__device__ __forceinline__ void cp_async16(const bf16* g, bf16* l) {
    __builtin_amdgcn_global_load_async_to_lds_b128((gv4i*)g, (lv4i*)l, 0, 0);
}
__device__ __forceinline__ void wait_async0() {
#if __has_builtin(__builtin_amdgcn_s_wait_asynccnt)
    __builtin_amdgcn_s_wait_asynccnt(0);
#else
    asm volatile("s_wait_asynccnt 0x0" ::: "memory");
#endif
}
#endif

// ---------------------------------------------------------------------------
// fp32 -> bf16 conversion
// ---------------------------------------------------------------------------
__global__ __launch_bounds__(BDIM)
void k_cvt_bf16(const float* __restrict__ in, bf16* __restrict__ out, int n) {
    int i = (blockIdx.x * BDIM + threadIdx.x) * 4;
    if (i + 3 < n) {
        float4 v = *(const float4*)(in + i);
        out[i + 0] = (bf16)v.x;
        out[i + 1] = (bf16)v.y;
        out[i + 2] = (bf16)v.z;
        out[i + 3] = (bf16)v.w;
    } else {
        for (int j = i; j < n; ++j) out[j] = (bf16)in[j];
    }
}

// ---------------------------------------------------------------------------
// Column L2 norms of Q and K over the channel dim: invq/invk[b*D + n]
// ---------------------------------------------------------------------------
__global__ __launch_bounds__(BDIM)
void k_colnorm(const bf16* __restrict__ Q, const bf16* __restrict__ K,
               float* __restrict__ invq, float* __restrict__ invk) {
    int t = blockIdx.x * BDIM + threadIdx.x;   // over BSZ*DDIM
    int b = t / DDIM, n = t % DDIM;
    const bf16* q = Q + (size_t)b * CDIM * DDIM + n;
    const bf16* k = K + (size_t)b * CDIM * DDIM + n;
    float sq = 0.f, sk = 0.f;
    for (int c = 0; c < CDIM; ++c) {
        float qv = (float)q[(size_t)c * DDIM];
        float kv = (float)k[(size_t)c * DDIM];
        sq += qv * qv;
        sk += kv * kv;
    }
    invq[t] = 1.0f / sqrtf(sq);
    invk[t] = 1.0f / sqrtf(sk);
}

// ---------------------------------------------------------------------------
// Per (b,c) row pass: scale K in place by invk[n] (-> Kn), row-sum Kn -> ksum,
// row-sum raw V -> vsum.  One block per (b,c).
// ---------------------------------------------------------------------------
__global__ __launch_bounds__(BDIM)
void k_rows(bf16* __restrict__ K, const bf16* __restrict__ V,
            const float* __restrict__ invk,
            float* __restrict__ ksum, float* __restrict__ vsum) {
    int c = blockIdx.x, b = blockIdx.y;
    size_t base = ((size_t)b * CDIM + c) * DDIM;
    const float* ik = invk + b * DDIM;
    float sk = 0.f, sv = 0.f;
    for (int n = threadIdx.x; n < DDIM; n += BDIM) {
        float kv = (float)K[base + n] * ik[n];
        K[base + n] = (bf16)kv;
        sk += kv;
        sv += (float)V[base + n];
    }
    #pragma unroll
    for (int o = 16; o > 0; o >>= 1) {
        sk += __shfl_down(sk, o, 32);
        sv += __shfl_down(sv, o, 32);
    }
    __shared__ float rk[BDIM / 32], rv[BDIM / 32];
    int lane = threadIdx.x & 31, w = threadIdx.x >> 5;
    if (lane == 0) { rk[w] = sk; rv[w] = sv; }
    __syncthreads();
    if (threadIdx.x == 0) {
        float a = 0.f, v2 = 0.f;
        for (int i = 0; i < BDIM / 32; ++i) { a += rk[i]; v2 += rv[i]; }
        ksum[b * CDIM + c] = a;
        vsum[b * CDIM + c] = v2;
    }
}

// ---------------------------------------------------------------------------
// Qt[b, n, c] = Q[b, c, n] * invq[b, n]   (normalized transpose, LDS tiled)
// ---------------------------------------------------------------------------
__global__ __launch_bounds__(BDIM)
void k_transposeQ(const bf16* __restrict__ Q, const float* __restrict__ invq,
                  bf16* __restrict__ Qt) {
    __shared__ bf16 t[32][33];
    int b = blockIdx.z;
    int n0 = blockIdx.x * 32, c0 = blockIdx.y * 32;
    const bf16* q = Q + (size_t)b * CDIM * DDIM;
    int tx = threadIdx.x & 31, ty = threadIdx.x >> 5;   // 8 rows of 32
    #pragma unroll
    for (int i = 0; i < 32; i += 8) {
        int c = c0 + ty + i, n = n0 + tx;
        t[ty + i][tx] = (bf16)((float)q[(size_t)c * DDIM + n] * invq[b * DDIM + n]);
    }
    __syncthreads();
    bf16* qt = Qt + (size_t)b * DDIM * CDIM;
    #pragma unroll
    for (int i = 0; i < 32; i += 8) {
        int n = n0 + ty + i, c = c0 + tx;
        qt[(size_t)n * CDIM + c] = t[tx][ty + i];
    }
}

// ---------------------------------------------------------------------------
// tailor[b,n] = 1 / (L + sum_c Qt[b,n,c] * (ksum[b,c] + EPS)); one wave/(b,n)
// ---------------------------------------------------------------------------
__global__ __launch_bounds__(BDIM)
void k_tailor(const bf16* __restrict__ Qt, const float* __restrict__ ksum,
              float* __restrict__ tailor) {
    int row = blockIdx.x * (BDIM / 32) + (threadIdx.x >> 5);  // over BSZ*DDIM
    int lane = threadIdx.x & 31;
    int b = row / DDIM;
    const bf16* q = Qt + (size_t)row * CDIM;
    const float* ks = ksum + b * CDIM;
    float s = 0.f;
    for (int c = lane; c < CDIM; c += 32)
        s += (float)q[c] * (ks[c] + 1e-6f);
    #pragma unroll
    for (int o = 16; o > 0; o >>= 1) s += __shfl_down(s, o, 32);
    if (lane == 0) tailor[row] = 1.0f / ((float)DDIM + s);
}

// ---------------------------------------------------------------------------
// Tiled BF16 WMMA GEMM:  Out[M,N] (+epilogue) = A(M,K) * Bt(N,K)^T
// Block tile 128x256, BK=32, double-buffered LDS via async global->LDS DMA,
// 8 waves -> 2(M) x 4(N); each wave 64x64 = 4x4 WMMA tiles.
// EPI 0: bf16 out + per-N fp32 bias      (projections)
// EPI 1: bf16 out                        (K*V^T matrix)
// EPI 2: fp32 out = X + g*(vsum+acc)*tl  (final result, fused epilogue)
// ---------------------------------------------------------------------------
template <int EPI>
__global__ __launch_bounds__(BDIM, 1)
void k_gemm(const bf16* __restrict__ Ag, const bf16* __restrict__ Bg,
            int M, int N, int K,
            size_t sA, size_t sB, size_t sO,
            void* __restrict__ Out,
            const float* __restrict__ bias,
            const float* __restrict__ Xres,
            const float* __restrict__ vsum,
            const float* __restrict__ tail,
            const float* __restrict__ gamma) {
    constexpr int BK = 32, LDT = 40;           // 80B row stride: conflict-free b128
    constexpr int BM = 128, BN = 256;
    __shared__ bf16 As[2][BM * LDT];           // 2 x 10 KB
    __shared__ bf16 Bs[2][BN * LDT];           // 2 x 20 KB

    int bz = blockIdx.z;
    const bf16* Ab = Ag + (size_t)bz * sA + (size_t)(blockIdx.y * BM) * K;
    const bf16* Bb = Bg + (size_t)bz * sB + (size_t)(blockIdx.x * BN) * K;

    int tid = threadIdx.x;
    int lane = tid & 31, wave = tid >> 5;
    int wm = wave >> 2, wn = wave & 3;         // 2 x 4 wave grid
    int lr = lane & 15, kh = (lane >> 4) * 8;  // WMMA 16-bit operand layout

    // cooperative staging slots (8 bf16 = 16B each)
    int ar[2], ac[2], br[4], bc[4];
    #pragma unroll
    for (int j = 0; j < 2; ++j) { int s = tid + j * BDIM; ar[j] = s >> 2; ac[j] = (s & 3) << 3; }
    #pragma unroll
    for (int j = 0; j < 4; ++j) { int s = tid + j * BDIM; br[j] = s >> 2; bc[j] = (s & 3) << 3; }

    v8f acc[16] = {};
    int nk = K / BK;

#if HAS_ASYNC_LDS
    // ---- preload K-tile 0 into buffer 0 via async DMA ----
    #pragma unroll
    for (int j = 0; j < 2; ++j)
        cp_async16(Ab + (size_t)ar[j] * K + ac[j], &As[0][ar[j] * LDT + ac[j]]);
    #pragma unroll
    for (int j = 0; j < 4; ++j)
        cp_async16(Bb + (size_t)br[j] * K + bc[j], &Bs[0][br[j] * LDT + bc[j]]);
    wait_async0();
    __syncthreads();
#else
    #pragma unroll
    for (int j = 0; j < 2; ++j)
        *(uint4*)&As[0][ar[j] * LDT + ac[j]] = *(const uint4*)(Ab + (size_t)ar[j] * K + ac[j]);
    #pragma unroll
    for (int j = 0; j < 4; ++j)
        *(uint4*)&Bs[0][br[j] * LDT + bc[j]] = *(const uint4*)(Bb + (size_t)br[j] * K + bc[j]);
    __syncthreads();
#endif

    for (int ki = 0; ki < nk; ++ki) {
        int cur = ki & 1;
        int nxt = cur ^ 1;
        bool has_next = (ki + 1 < nk);

#if HAS_ASYNC_LDS
        // ---- issue async DMA for tile ki+1 (no data VGPRs held) ----
        if (has_next) {
            int k0 = (ki + 1) * BK;
            #pragma unroll
            for (int j = 0; j < 2; ++j)
                cp_async16(Ab + (size_t)ar[j] * K + k0 + ac[j],
                           &As[nxt][ar[j] * LDT + ac[j]]);
            #pragma unroll
            for (int j = 0; j < 4; ++j)
                cp_async16(Bb + (size_t)br[j] * K + k0 + bc[j],
                           &Bs[nxt][br[j] * LDT + bc[j]]);
        }
#else
        uint4 ra[2], rb[4];
        if (has_next) {
            int k0 = (ki + 1) * BK;
            #pragma unroll
            for (int j = 0; j < 2; ++j)
                ra[j] = *(const uint4*)(Ab + (size_t)ar[j] * K + k0 + ac[j]);
            #pragma unroll
            for (int j = 0; j < 4; ++j)
                rb[j] = *(const uint4*)(Bb + (size_t)br[j] * K + k0 + bc[j]);
        }
#endif

        // ---- fragments from LDS (two b128 per fragment per lane) ----
        v16bf bfm[4];
        #pragma unroll
        for (int t = 0; t < 4; ++t) {
            const bf16* p = &Bs[cur][(wn * 64 + t * 16 + lr) * LDT + kh];
            uint4* d = (uint4*)&bfm[t];
            d[0] = *(const uint4*)p;
            d[1] = *(const uint4*)(p + 16);
        }
        #pragma unroll
        for (int tm = 0; tm < 4; ++tm) {
            v16bf af;
            const bf16* p = &As[cur][(wm * 64 + tm * 16 + lr) * LDT + kh];
            uint4* d = (uint4*)&af;
            d[0] = *(const uint4*)p;
            d[1] = *(const uint4*)(p + 16);
            #pragma unroll
            for (int tn = 0; tn < 4; ++tn)
                acc[tm * 4 + tn] = __builtin_amdgcn_wmma_f32_16x16x32_bf16(
                    false, af, false, bfm[tn], (short)0, acc[tm * 4 + tn],
                    false, false);
        }

#if HAS_ASYNC_LDS
        if (has_next) wait_async0();
#else
        if (has_next) {
            #pragma unroll
            for (int j = 0; j < 2; ++j) *(uint4*)&As[nxt][ar[j] * LDT + ac[j]] = ra[j];
            #pragma unroll
            for (int j = 0; j < 4; ++j) *(uint4*)&Bs[nxt][br[j] * LDT + bc[j]] = rb[j];
        }
#endif
        __syncthreads();
    }

    // ---- epilogue: C/D layout: VGPR r -> row r + 8*(lane>>4), col lane&15 ----
    int row0 = blockIdx.y * BM + wm * 64 + ((lane >> 4) * 8);
    int col0 = blockIdx.x * BN + wn * 64 + (lane & 15);

    float g = 0.f;
    if constexpr (EPI == 2) g = gamma[0];

    #pragma unroll
    for (int tm = 0; tm < 4; ++tm) {
        #pragma unroll
        for (int tn = 0; tn < 4; ++tn) {
            v8f a = acc[tm * 4 + tn];
            int row = row0 + tm * 16;
            int col = col0 + tn * 16;
            #pragma unroll
            for (int r = 0; r < 8; ++r) {
                size_t idx = (size_t)(row + r) * N + col;
                if constexpr (EPI == 0) {
                    ((bf16*)Out + (size_t)bz * sO)[idx] = (bf16)(a[r] + bias[col]);
                } else if constexpr (EPI == 1) {
                    ((bf16*)Out + (size_t)bz * sO)[idx] = (bf16)a[r];
                } else {
                    const float* X = Xres + (size_t)bz * sO;
                    float v = X[idx] +
                              g * (vsum[(size_t)bz * M + row + r] + a[r]) *
                                  tail[(size_t)bz * N + col];
                    ((float*)Out + (size_t)bz * sO)[idx] = v;
                }
            }
        }
    }
}

// ---------------------------------------------------------------------------
extern "C" void kernel_launch(void* const* d_in, const int* in_sizes, int n_in,
                              void* d_out, int out_size, void* d_ws, size_t ws_size,
                              hipStream_t stream) {
    (void)in_sizes; (void)n_in; (void)out_size; (void)ws_size;
    const float* x     = (const float*)d_in[0];
    const float* Wq    = (const float*)d_in[1];
    const float* bq    = (const float*)d_in[2];
    const float* Wk    = (const float*)d_in[3];
    const float* bk    = (const float*)d_in[4];
    const float* Wv    = (const float*)d_in[5];
    const float* bv    = (const float*)d_in[6];
    const float* gamma = (const float*)d_in[7];
    float* out = (float*)d_out;

    const size_t nBCD = (size_t)BSZ * CDIM * DDIM;   // 33,554,432
    const size_t nW   = (size_t)DDIM * DDIM;         // 1,048,576

    char* ws = (char*)d_ws;
    size_t off = 0;
    auto alloc = [&](size_t bytes) -> void* {
        void* p = ws + off;
        off += (bytes + 255) & ~(size_t)255;
        return p;
    };
    bf16* xb   = (bf16*)alloc(nBCD * 2);
    bf16* Wb   = (bf16*)alloc(3 * nW * 2);
    bf16* Qb   = (bf16*)alloc(nBCD * 2);
    bf16* Kb   = (bf16*)alloc(nBCD * 2);
    bf16* Vb   = (bf16*)alloc(nBCD * 2);
    bf16* Qt   = (bf16*)alloc(nBCD * 2);
    bf16* MT   = (bf16*)alloc((size_t)BSZ * CDIM * CDIM * 2);   // 128 MB
    float* invq   = (float*)alloc((size_t)BSZ * DDIM * 4);
    float* invk   = (float*)alloc((size_t)BSZ * DDIM * 4);
    float* ksum   = (float*)alloc((size_t)BSZ * CDIM * 4);
    float* vsum   = (float*)alloc((size_t)BSZ * CDIM * 4);
    float* tailor = (float*)alloc((size_t)BSZ * DDIM * 4);

    // --- fp32 -> bf16 conversions ---
    k_cvt_bf16<<<(int)(nBCD / 4 / BDIM), BDIM, 0, stream>>>(x, xb, (int)nBCD);
    k_cvt_bf16<<<(int)(nW / 4 / BDIM), BDIM, 0, stream>>>(Wq, Wb + 0 * nW, (int)nW);
    k_cvt_bf16<<<(int)(nW / 4 / BDIM), BDIM, 0, stream>>>(Wk, Wb + 1 * nW, (int)nW);
    k_cvt_bf16<<<(int)(nW / 4 / BDIM), BDIM, 0, stream>>>(Wv, Wb + 2 * nW, (int)nW);

    // --- projections: Q/K/V[b] = x[b] @ W^T + b  (M=C, N=D, K=D) ---
    dim3 gp(DDIM / 256, CDIM / 128, BSZ);
    k_gemm<0><<<gp, BDIM, 0, stream>>>(xb, Wb + 0 * nW, CDIM, DDIM, DDIM,
                                       nBCD / BSZ, 0, nBCD / BSZ,
                                       Qb, bq, nullptr, nullptr, nullptr, nullptr);
    k_gemm<0><<<gp, BDIM, 0, stream>>>(xb, Wb + 1 * nW, CDIM, DDIM, DDIM,
                                       nBCD / BSZ, 0, nBCD / BSZ,
                                       Kb, bk, nullptr, nullptr, nullptr, nullptr);
    k_gemm<0><<<gp, BDIM, 0, stream>>>(xb, Wb + 2 * nW, CDIM, DDIM, DDIM,
                                       nBCD / BSZ, 0, nBCD / BSZ,
                                       Vb, bv, nullptr, nullptr, nullptr, nullptr);

    // --- column norms, K scale + row sums, Q transpose, tailor ---
    k_colnorm<<<(BSZ * DDIM) / BDIM, BDIM, 0, stream>>>(Qb, Kb, invq, invk);
    k_rows<<<dim3(CDIM, BSZ), BDIM, 0, stream>>>(Kb, Vb, invk, ksum, vsum);
    k_transposeQ<<<dim3(DDIM / 32, CDIM / 32, BSZ), BDIM, 0, stream>>>(Qb, invq, Qt);
    k_tailor<<<(BSZ * DDIM) / (BDIM / 32), BDIM, 0, stream>>>(Qt, ksum, tailor);

    // --- MT[b][c,m] = sum_n V[b,c,n] * Kn[b,m,n]   (M=C, N=C, K=D) ---
    dim3 gkv(CDIM / 256, CDIM / 128, BSZ);
    k_gemm<1><<<gkv, BDIM, 0, stream>>>(Vb, Kb, CDIM, CDIM, DDIM,
                                        nBCD / BSZ, nBCD / BSZ,
                                        (size_t)CDIM * CDIM,
                                        MT, nullptr, nullptr, nullptr, nullptr, nullptr);

    // --- out[b,c,n] = x + gamma*(vsum[c] + sum_m MT[c,m]*Qt[n,m]) * tailor[n] ---
    dim3 go(DDIM / 256, CDIM / 128, BSZ);
    k_gemm<2><<<go, BDIM, 0, stream>>>(MT, Qt, CDIM, DDIM, CDIM,
                                       (size_t)CDIM * CDIM, nBCD / BSZ, nBCD / BSZ,
                                       out, nullptr, x, vsum, tailor, gamma);
}
